// GlobalAttention_62637803045228
// MI455X (gfx1250) — matrix operand order repaired and checked by reading, not compiled, not run on hardware
//
#include <hip/hip_runtime.h>
#include <hip/hip_bf16.h>
#include <math.h>

typedef __attribute__((ext_vector_type(16))) __bf16 v16bf;
typedef __attribute__((ext_vector_type(8)))  float  v8f;

#define N_NODES_C  1000000
#define N_GRAPHS_C 4096

#define LOG2E 1.4426950408889634f
#define LN2   0.6931471805599453f

// ---- order-preserving float <-> uint map for atomicMax on floats ----
__device__ __forceinline__ unsigned f2ord(float f) {
    unsigned u = __float_as_uint(f);
    return (u & 0x80000000u) ? ~u : (u | 0x80000000u);
}
__device__ __forceinline__ float ord2f(unsigned u) {
    return (u & 0x80000000u) ? __uint_as_float(u & 0x7fffffffu) : __uint_as_float(~u);
}

// stable softplus via hardware v_exp_f32 / v_log_f32 (base-2 transcendentals)
__device__ __forceinline__ float softplus_fast(float v) {
    float t = __builtin_amdgcn_exp2f(-fabsf(v) * LOG2E);   // exp(-|v|) in (0,1]
    return fmaxf(v, 0.0f) + LN2 * __builtin_amdgcn_logf(1.0f + t);
}

// ---- kernel 1: g = global_attr @ Wg + bg  (4096 x 64, trivial) ----
__global__ void __launch_bounds__(256)
ga_prep_g(const float* __restrict__ ga, const float* __restrict__ Wg,
          const float* __restrict__ bg, float* __restrict__ g) {
    int idx = blockIdx.x * 256 + threadIdx.x;
    if (idx >= N_GRAPHS_C * 64) return;
    int b = idx >> 6, j = idx & 63;
    float acc = bg[j];
    acc += ga[b * 3 + 0] * Wg[0 * 64 + j];
    acc += ga[b * 3 + 1] * Wg[1 * 64 + j];
    acc += ga[b * 3 + 2] * Wg[2 * 64 + j];
    g[idx] = acc;
}

// ---- kernel 2: swizzle Wn[128][64] (f32) into bf16 WMMA B-fragments,
//      and (re)initialize per-segment max/sum accumulators ----
// frag index = (((c*4 + t)*32 + lane)*16 + e), where B element (k, n):
//   k = c*32 + (lane>>4)*8 + (e&7) + (e>>3)*16,  n = t*16 + (lane&15)
__global__ void __launch_bounds__(256)
ga_prep_wn(const float* __restrict__ Wn, __bf16* __restrict__ frag,
           unsigned* __restrict__ segmax, float* __restrict__ segsum) {
    int idx = blockIdx.x * 256 + threadIdx.x;
    if (idx < 8192) {
        int e    = idx & 15;
        int lane = (idx >> 4) & 31;
        int t    = (idx >> 9) & 3;
        int c    = (idx >> 11) & 3;
        int k    = c * 32 + (lane >> 4) * 8 + (e & 7) + (e >> 3) * 16;
        int col  = t * 16 + (lane & 15);
        frag[idx] = (__bf16)Wn[k * 64 + col];
    }
    if (idx < N_GRAPHS_C) {
        segmax[idx] = f2ord(-INFINITY);
        segsum[idx] = 0.0f;
    }
}

// ---- kernel 3: fused per-16-node tile:
//  h = softplus([x | g[seg]] @ Wn + bn); s = h @ Wa + ba; atomicMax segment max.
//  One wave per 16-node tile. 4 K-chunks x 4 N-tiles = 16 WMMAs. ----
__global__ void __launch_bounds__(256)
ga_main(const float* __restrict__ x, const int* __restrict__ nb,
        const float* __restrict__ g, const __bf16* __restrict__ wnfrag,
        const float* __restrict__ bn, const float* __restrict__ Wa,
        const float* __restrict__ ba_p, float* __restrict__ s_out,
        unsigned* __restrict__ segmax, int n_tiles) {
    __shared__ v16bf bsh[512];  // 16 KB: full Wn fragment table
    {
        const v16bf* wf = (const v16bf*)wnfrag;
        for (int i = threadIdx.x; i < 512; i += 256) bsh[i] = wf[i];
    }
    __syncthreads();

    int lane  = threadIdx.x & 31;
    int wtile = (int)((blockIdx.x * 256u + threadIdx.x) >> 5);
    if (wtile >= n_tiles) return;  // wave-uniform guard

    int half     = lane >> 4;
    int rrow     = lane & 15;
    int tilebase = wtile * 16;
    int row      = tilebase + rrow;

    int sg = nb[row];
    const float* xr = x + (size_t)row * 64;
    const float* gr = g + (size_t)sg * 64;

    // A fragments: lane holds row M=rrow; elems 0..7 = K=(half*8)+0..7,
    // elems 8..15 = same +16 (16-bit A 16x32 VGPR layout). Vector loads.
    v16bf a[4];
#pragma unroll
    for (int c = 0; c < 4; ++c) {
        const float* src = (c < 2) ? xr : gr;
        int base = ((c & 1) * 32) + half * 8;
        float4 lo0 = *(const float4*)(src + base);
        float4 lo1 = *(const float4*)(src + base + 4);
        float4 hi0 = *(const float4*)(src + base + 16);
        float4 hi1 = *(const float4*)(src + base + 20);
        a[c][0]  = (__bf16)lo0.x; a[c][1]  = (__bf16)lo0.y;
        a[c][2]  = (__bf16)lo0.z; a[c][3]  = (__bf16)lo0.w;
        a[c][4]  = (__bf16)lo1.x; a[c][5]  = (__bf16)lo1.y;
        a[c][6]  = (__bf16)lo1.z; a[c][7]  = (__bf16)lo1.w;
        a[c][8]  = (__bf16)hi0.x; a[c][9]  = (__bf16)hi0.y;
        a[c][10] = (__bf16)hi0.z; a[c][11] = (__bf16)hi0.w;
        a[c][12] = (__bf16)hi1.x; a[c][13] = (__bf16)hi1.y;
        a[c][14] = (__bf16)hi1.z; a[c][15] = (__bf16)hi1.w;
    }

    // Fold bn into the C-matrix: in the 16x16 f32 C/D layout every register r
    // of a lane shares the same column n, so C[r] = bn[n] is the exact bias.
    float bnv[4], wav[4];
#pragma unroll
    for (int t = 0; t < 4; ++t) {
        int n = t * 16 + rrow;
        bnv[t] = bn[n];
        wav[t] = Wa[n];
    }

    v8f acc[4];
#pragma unroll
    for (int t = 0; t < 4; ++t)
#pragma unroll
        for (int r = 0; r < 8; ++r) acc[t][r] = bnv[t];

#pragma unroll
    for (int t = 0; t < 4; ++t) {
#pragma unroll
        for (int c = 0; c < 4; ++c) {
            v16bf b = bsh[(c * 4 + t) * 32 + lane];
            acc[t] = __builtin_amdgcn_wmma_f32_16x16x32_bf16(
                false, a[c], false, b, (short)0, acc[t], false, false);
        }
    }

    // Epilogue: softplus then dot with Wa. Lane holds n = rrow (within tile t),
    // register r holds m = r + 8*half.
    float sp[8];
#pragma unroll
    for (int r = 0; r < 8; ++r) sp[r] = 0.0f;
#pragma unroll
    for (int t = 0; t < 4; ++t) {
#pragma unroll
        for (int r = 0; r < 8; ++r) {
            sp[r] += softplus_fast(acc[t][r]) * wav[t];
        }
    }
    // Reduce the n-dimension across each 16-lane half (xor masks keep halves).
#pragma unroll
    for (int off = 1; off < 16; off <<= 1) {
#pragma unroll
        for (int r = 0; r < 8; ++r) sp[r] += __shfl_xor(sp[r], off, 32);
    }
    float ba = ba_p[0];
    if (rrow == 0) {
#pragma unroll
        for (int r = 0; r < 8; ++r) {
            int node = tilebase + half * 8 + r;
            float sv = sp[r] + ba;
            s_out[node] = sv;
            atomicMax(&segmax[nb[node]], f2ord(sv));
        }
    }
}

// ---- kernel 4: e = exp(s - max[seg]); segment sums via atomicAdd ----
__global__ void __launch_bounds__(256)
ga_exp(const float* __restrict__ s, const int* __restrict__ nb,
       const unsigned* __restrict__ segmax, float* __restrict__ e,
       float* __restrict__ segsum, int n) {
    int i = blockIdx.x * 256 + threadIdx.x;
    if (i >= n) return;
    int sg = nb[i];
    float ev = __builtin_amdgcn_exp2f((s[i] - ord2f(segmax[sg])) * LOG2E);
    e[i] = ev;
    atomicAdd(&segsum[sg], ev);
}

// ---- kernel 5: w = e / (sum[seg] + eps) ----
__global__ void __launch_bounds__(256)
ga_norm(const float* __restrict__ e, const int* __restrict__ nb,
        const float* __restrict__ segsum, float* __restrict__ w, int n) {
    int i = blockIdx.x * 256 + threadIdx.x;
    if (i >= n) return;
    w[i] = e[i] / (segsum[nb[i]] + 1e-16f);
}

extern "C" void kernel_launch(void* const* d_in, const int* in_sizes, int n_in,
                              void* d_out, int out_size, void* d_ws, size_t ws_size,
                              hipStream_t stream) {
    const float* x   = (const float*)d_in[0];   // [N,64]
    const int*   nb  = (const int*)d_in[1];     // [N]
    const float* ga  = (const float*)d_in[2];   // [B,3]
    const float* Wg  = (const float*)d_in[3];   // [3,64]
    const float* bg  = (const float*)d_in[4];   // [64]
    const float* Wn  = (const float*)d_in[5];   // [128,64]
    const float* bn  = (const float*)d_in[6];   // [64]
    const float* Wa  = (const float*)d_in[7];   // [64,1]
    const float* ba  = (const float*)d_in[8];   // [1]
    float* out = (float*)d_out;

    const int N = in_sizes[0] / 64;             // 1,000,000
    const int n_tiles = N / 16;                 // 62,500

    // workspace layout
    char* ws = (char*)d_ws;
    float*    g      = (float*)(ws);                         // 4096*64 f32 = 1 MB
    __bf16*   frag   = (__bf16*)(ws + 1048576);              // 8192 bf16  = 16 KB
    unsigned* segmax = (unsigned*)(ws + 1048576 + 16384);    // 4096 u32
    float*    segsum = (float*)(ws + 1048576 + 32768);       // 4096 f32
    float*    s      = (float*)(ws + 1048576 + 49152);       // N f32
    float*    e      = s + N;                                // N f32

    ga_prep_g<<<(N_GRAPHS_C * 64 + 255) / 256, 256, 0, stream>>>(ga, Wg, bg, g);
    ga_prep_wn<<<32, 256, 0, stream>>>(Wn, frag, segmax, segsum);

    int main_blocks = (n_tiles * 32 + 255) / 256;
    ga_main<<<main_blocks, 256, 0, stream>>>(x, nb, g, frag, bn, Wa, ba,
                                             s, segmax, n_tiles);

    int nblk = (N + 255) / 256;
    ga_exp<<<nblk, 256, 0, stream>>>(s, nb, segmax, e, segsum, N);
    ga_norm<<<nblk, 256, 0, stream>>>(e, nb, segsum, out, N);
}
